// FleximodalFuseMoE_45114336477546
// MI455X (gfx1250) — matrix operand order repaired and checked by reading, not compiled, not run on hardware
//
#include <hip/hip_runtime.h>
#include <hip/hip_bf16.h>
#include <math.h>

// ---------------- problem sizes (from reference) ----------------
constexpr int N_TOK   = 4 * 1024;   // B*T
constexpr int D_MODEL = 1024;
constexpr int N_EXP   = 8;
constexpr int TOPK    = 2;
constexpr int D_FF    = 4096;
constexpr int N_ASG   = N_TOK * TOPK;   // 8192 assignments

// ---------------- GEMM tiling ----------------
constexpr int TM  = 128;  // tokens per tile
constexpr int TN  = 128;  // output cols per tile
constexpr int TKC = 64;   // K halves staged per buffer (2 WMMA k-steps)
constexpr int LDW = 72;   // LDS row pitch in halves (144B = 32 DW + 4 DW pad; TDM pad-compatible)

typedef __bf16 bf16;
typedef bf16  v16bf __attribute__((ext_vector_type(16)));
typedef bf16  v8bf  __attribute__((ext_vector_type(8)));
typedef float v8f   __attribute__((ext_vector_type(8)));
typedef float f32x4 __attribute__((ext_vector_type(4)));
typedef unsigned int u32;
typedef u32 u32x4 __attribute__((ext_vector_type(4)));
typedef u32 u32x8 __attribute__((ext_vector_type(8)));

#define USE_ASYNC_LDS 1
#define USE_TDM 1

// ---------------- CDNA5 async copy to LDS (per-lane 16B, ASYNCcnt) ----------------
__device__ __forceinline__ void cp16_async(void* lds_generic, const void* gsrc) {
#if USE_ASYNC_LDS
  u32 lds_byte = (u32)(size_t)lds_generic; // generic LDS addr low 32 bits == LDS offset
  asm volatile("global_load_async_to_lds_b128 %0, %1, off"
               :: "v"(lds_byte), "v"(gsrc) : "memory");
#else
  *(f32x4*)lds_generic = *(const f32x4*)gsrc;
#endif
}
__device__ __forceinline__ void wait_async_lds() {
#if USE_ASYNC_LDS
  asm volatile("s_wait_asynccnt 0" ::: "memory");
#endif
}

// ---------------- CDNA5 Tensor Data Mover: 2D tile global->LDS ----------------
// 128 rows x 64 halves (128B row), row pitch `srcPitchHalves`; LDS gets 16B pad
// after each 128B row (pad_interval=32 DW code 4, pad_amount=4 DW code 3) => LDW=72.
__device__ __forceinline__ void tdm_load_tile_2d(void* lds_dst, const bf16* gsrc,
                                                 u32 srcPitchHalves) {
#if USE_TDM
  unsigned long long ga = (unsigned long long)(size_t)gsrc;
  u32x4 g0;
  g0[0] = 1u;                                   // count=1, no gather, user mode
  g0[1] = (u32)(size_t)lds_dst;                 // lds_addr (bytes)
  g0[2] = (u32)ga;                              // global_addr[31:0]
  g0[3] = (u32)((ga >> 32) & 0x1FFFFFFull) | (2u << 30);  // addr[56:32] | type=2
  u32x8 g1;
  g1[0] = (1u << 16)      // data_size = 2 bytes
        | (1u << 20)      // pad_enable
        | (4u << 22)      // pad_interval: 32 DWORDs (one 128B row)
        | (3u << 25);     // pad_amount:   4 DWORDs (16B)
  g1[1] = ((u32)D_FF & 0xFFFFu) << 16;          // tensor_dim0[15:0] (halves)
  g1[2] = (((u32)D_FF >> 16) & 0xFFFFu) | (((u32)N_ASG & 0xFFFFu) << 16); // dim0 hi | dim1 lo
  g1[3] = (((u32)N_ASG >> 16) & 0xFFFFu) | ((u32)TKC << 16);  // dim1 hi | tile_dim0=64 halves
  g1[4] = (u32)TM;                              // tile_dim1=128 rows, tile_dim2=0 (2D)
  g1[5] = srcPitchHalves;                       // tensor_dim0_stride[31:0] (halves)
  g1[6] = 0u;
  g1[7] = 0u;
  asm volatile("tensor_load_to_lds %0, %1" :: "s"(g0), "s"(g1) : "memory");
#else
  (void)lds_dst; (void)gsrc; (void)srcPitchHalves;
#endif
}
__device__ __forceinline__ void wait_tensor() {
#if USE_TDM
  asm volatile("s_wait_tensorcnt 0x0" ::: "memory");
#endif
}

__device__ __forceinline__ float gelu_exact(float v) {
  return 0.5f * v * (1.0f + erff(v * 0.70710678118654752f));
}

// ================================================================
// K0: zero expert counters
// ================================================================
__global__ void moe_zero_meta(int* cnt) {
  if (threadIdx.x < N_EXP) cnt[threadIdx.x] = 0;
}

// ================================================================
// K1: fused LayerNorm + router (logits, top-2, softmax gates)
// ================================================================
__global__ __launch_bounds__(256) void moe_ln_router(
    const float* __restrict__ x, const float* __restrict__ ln_g,
    const float* __restrict__ ln_b, const float* __restrict__ Wr,
    const float* __restrict__ br,
    bf16* __restrict__ hbf, int* __restrict__ idxo, float* __restrict__ gateo,
    int* __restrict__ cnt) {
  const int t   = blockIdx.x;
  const int tid = threadIdx.x;
  const int wave = tid >> 5, lane = tid & 31;
  const float* xr = x + (size_t)t * D_MODEL;

  __shared__ float red[16];
  __shared__ float pr[8][8];

  const int d0 = tid * 4;
  f32x4 xv = *(const f32x4*)(xr + d0);
  float xa[4] = {xv[0], xv[1], xv[2], xv[3]};

  float s = xa[0] + xa[1] + xa[2] + xa[3];
  float s2 = xa[0]*xa[0] + xa[1]*xa[1] + xa[2]*xa[2] + xa[3]*xa[3];
#pragma unroll
  for (int o = 16; o > 0; o >>= 1) {
    s  += __shfl_down(s,  o, 32);
    s2 += __shfl_down(s2, o, 32);
  }
  if (lane == 0) { red[wave] = s; red[8 + wave] = s2; }
  __syncthreads();
  if (tid == 0) {
    float a = 0.f, b = 0.f;
#pragma unroll
    for (int i = 0; i < 8; ++i) { a += red[i]; b += red[8 + i]; }
    float mu = a / (float)D_MODEL;
    red[0] = mu;
    red[1] = b / (float)D_MODEL - mu * mu;
  }
  __syncthreads();
  const float mu = red[0];
  const float rstd = rsqrtf(red[1] + 1e-5f);

  float p[8];
#pragma unroll
  for (int e = 0; e < 8; ++e) p[e] = 0.f;
#pragma unroll
  for (int j = 0; j < 4; ++j) {
    const int d = d0 + j;
    float hn = (xa[j] - mu) * rstd * ln_g[d] + ln_b[d];
    hbf[(size_t)t * D_MODEL + d] = (bf16)hn;
    const float* wr = Wr + (size_t)d * N_EXP;
#pragma unroll
    for (int e = 0; e < 8; ++e) p[e] += hn * wr[e];
  }
#pragma unroll
  for (int e = 0; e < 8; ++e) {
#pragma unroll
    for (int o = 16; o > 0; o >>= 1) p[e] += __shfl_down(p[e], o, 32);
  }
  if (lane == 0) {
#pragma unroll
    for (int e = 0; e < 8; ++e) pr[wave][e] = p[e];
  }
  __syncthreads();
  if (tid == 0) {
    float lg[8];
#pragma unroll
    for (int e = 0; e < 8; ++e) {
      float a = br[e];
#pragma unroll
      for (int w = 0; w < 8; ++w) a += pr[w][e];
      lg[e] = a;
    }
    int i0 = 0;
#pragma unroll
    for (int e = 1; e < 8; ++e) if (lg[e] > lg[i0]) i0 = e;
    int i1 = -1;
#pragma unroll
    for (int e = 0; e < 8; ++e) {
      if (e == i0) continue;
      if (i1 < 0 || lg[e] > lg[i1]) i1 = e;
    }
    float e1 = __expf(lg[i1] - lg[i0]);
    float inv = 1.f / (1.f + e1);
    idxo[t * 2 + 0] = i0;  gateo[t * 2 + 0] = inv;
    idxo[t * 2 + 1] = i1;  gateo[t * 2 + 1] = e1 * inv;
    atomicAdd(&cnt[i0], 1);
    atomicAdd(&cnt[i1], 1);
  }
}

// ================================================================
// K2: exclusive prefix over 8 expert counts
// ================================================================
__global__ void moe_seg_offsets(const int* __restrict__ cnt,
                                int* __restrict__ off, int* __restrict__ cursor) {
  if (threadIdx.x == 0) {
    int a = 0;
    for (int e = 0; e < N_EXP; ++e) { off[e] = a; cursor[e] = a; a += cnt[e]; }
  }
}

// ================================================================
// K3: scatter (token, gate, slot) into per-expert segments
// ================================================================
__global__ __launch_bounds__(256) void moe_scatter(
    const int* __restrict__ idxo, const float* __restrict__ gateo,
    int* __restrict__ cursor, int* __restrict__ tok,
    float* __restrict__ ga, int* __restrict__ slot) {
  int t = blockIdx.x * 256 + threadIdx.x;
  if (t >= N_TOK) return;
#pragma unroll
  for (int s = 0; s < TOPK; ++s) {
    int e = idxo[t * 2 + s];
    int pos = atomicAdd(&cursor[e], 1);
    tok[pos] = t;
    ga[pos] = gateo[t * 2 + s];
    slot[pos] = s;
  }
}

// ================================================================
// K4: grouped GEMM1 -- h1 = gelu(h[tok] * W1[e] + b1[e])  (bf16 out)
// 64-K chunks, double-buffered LDS, 16 WMMA per wave per sync.
// A (gathered rows) via per-lane async-LDS DMA with HOISTED base addrs.
// B (fp32 weights) NT-loaded, pair-packed to bf16, transposed in LDS.
// ================================================================
__global__ __launch_bounds__(256) void moe_gemm1(
    const bf16* __restrict__ hbf, const float* __restrict__ W1,
    const float* __restrict__ b1,
    const int* __restrict__ segoff, const int* __restrict__ segcnt,
    const int* __restrict__ tok, bf16* __restrict__ h1) {
  __shared__ bf16 As[2][TM * LDW];
  __shared__ bf16 Bs[2][TN * LDW];

  const int e = blockIdx.z;
  const int cnt = segcnt[e];
  const int m0 = blockIdx.y * TM;
  if (m0 >= cnt) return;
  const int f0  = blockIdx.x * TN;
  const int off = segoff[e];
  const int tid = threadIdx.x;
  const int wave = tid >> 5, lane = tid & 31;
  const int wm = wave >> 1, wn = wave & 1;
  const int l15 = lane & 15, hi = lane >> 4;

  const float* W1e = W1 + (size_t)e * D_MODEL * D_FF;

  v8f acc[2][4];
#pragma unroll
  for (int a = 0; a < 2; ++a)
#pragma unroll
    for (int n = 0; n < 4; ++n)
#pragma unroll
      for (int q = 0; q < 8; ++q) acc[a][n][q] = 0.f;

  // ---- hoisted staging addresses (K-invariant) ----
  const bf16* srcA[4]; u32 dstA[4];          // A: 1024 16B chunks, 4/thread
#pragma unroll
  for (int i = 0; i < 4; ++i) {
    int idx = i * 256 + tid;
    int row = idx >> 3;                      // 0..127
    int seg = idx & 7;                       // 16B chunk in 128B row
    int rr = m0 + row;
    int tkn = tok[off + (rr < cnt ? rr : 0)];
    srcA[i] = hbf + (size_t)tkn * D_MODEL + seg * 8;
    dstA[i] = row * LDW + seg * 8;
  }
  const float* srcB[4]; u32 dstB[4];         // B: (k-pair 32) x (f4 32), 4/thread
#pragma unroll
  for (int i = 0; i < 4; ++i) {
    int idx = i * 256 + tid;
    int kp = idx >> 5;                       // 0..31 (k pair)
    int c4 = idx & 31;                       // 0..31 (4-col group)
    srcB[i] = W1e + (size_t)(2 * kp) * D_FF + f0 + c4 * 4;
    dstB[i] = (c4 * 4) * LDW + 2 * kp;
  }

  auto stageA = [&](int buf, int k0) {
#pragma unroll
    for (int i = 0; i < 4; ++i)
      cp16_async(&As[buf][dstA[i]], srcA[i] + k0);
  };
  auto stageB = [&](int buf, int k0) {
#pragma unroll
    for (int i = 0; i < 4; ++i) {
      const float* gp = srcB[i] + (size_t)k0 * D_FF;
      if (k0 + 2 * TKC < D_MODEL)
        __builtin_prefetch(gp + 2 * (size_t)TKC * D_FF, 0, 1);
      f32x4 wa = __builtin_nontemporal_load((const f32x4*)gp);
      f32x4 wb = __builtin_nontemporal_load((const f32x4*)(gp + D_FF));
#pragma unroll
      for (int j = 0; j < 4; ++j) {
        union { u32 u; bf16 h[2]; } pk;
        pk.h[0] = (bf16)wa[j];
        pk.h[1] = (bf16)wb[j];
        *(u32*)&Bs[buf][dstB[i] + j * LDW] = pk.u;   // 4B aligned
      }
    }
  };

  stageA(0, 0);
  stageB(0, 0);
  constexpr int NIT = D_MODEL / TKC;         // 16
  for (int it = 0; it < NIT; ++it) {
    const int cur = it & 1;
    wait_async_lds();
    __syncthreads();                         // buf `cur` fully staged by all waves
    if (it + 1 < NIT) {
      stageA(1 - cur, (it + 1) * TKC);
      stageB(1 - cur, (it + 1) * TKC);
    }
#pragma unroll
    for (int kk = 0; kk < TKC; kk += 32) {   // two wmma k-steps per chunk
      v16bf afrag[2];
#pragma unroll
      for (int a = 0; a < 2; ++a) {
        int row = (wm * 2 + a) * 16 + l15;
        int base = kk + (hi ? 8 : 0);
        const bf16* p = &As[cur][row * LDW];
        ((v8bf*)&afrag[a])[0] = *(const v8bf*)(p + base);
        ((v8bf*)&afrag[a])[1] = *(const v8bf*)(p + base + 16);
      }
      v16bf bfrag[4];
#pragma unroll
      for (int n = 0; n < 4; ++n) {
        int frow = (wn * 4 + n) * 16 + l15;
        int base = kk + (hi ? 8 : 0);
        const bf16* p = &Bs[cur][frow * LDW] + base;
        ((v8bf*)&bfrag[n])[0] = *(const v8bf*)(p);
        ((v8bf*)&bfrag[n])[1] = *(const v8bf*)(p + 16);
      }
#pragma unroll
      for (int a = 0; a < 2; ++a)
#pragma unroll
        for (int n = 0; n < 4; ++n)
          acc[a][n] = __builtin_amdgcn_wmma_f32_16x16x32_bf16(
              false, afrag[a], false, bfrag[n], (short)0, acc[a][n], false, false);
    }
  }

  const float* b1e = b1 + (size_t)e * D_FF;
#pragma unroll
  for (int a = 0; a < 2; ++a) {
#pragma unroll
    for (int n = 0; n < 4; ++n) {
      int col = f0 + (wn * 4 + n) * 16 + l15;
      float bv = b1e[col];
#pragma unroll
      for (int r = 0; r < 8; ++r) {
        int rowl = (wm * 2 + a) * 16 + (hi ? r + 8 : r);
        int grow = m0 + rowl;
        if (grow < cnt) {
          float v = gelu_exact(acc[a][n][r] + bv);
          h1[(size_t)(off + grow) * D_FF + col] = (bf16)v;
        }
      }
    }
  }
}

// ================================================================
// K5: grouped GEMM2 -- contrib[(tok,slot)] = (h1 * W2[e] + b2[e]) * gate
// A tile = regular 2D tile of h1 -> one Tensor Data Mover DMA per chunk
// (wave 0 issues; TDM deposits with padded 144B LDS pitch).
// ================================================================
__global__ __launch_bounds__(256) void moe_gemm2(
    const bf16* __restrict__ h1, const float* __restrict__ W2,
    const float* __restrict__ b2,
    const int* __restrict__ segoff, const int* __restrict__ segcnt,
    const int* __restrict__ tok, const float* __restrict__ ga,
    const int* __restrict__ slot, float* __restrict__ contrib) {
  __shared__ bf16 As[2][TM * LDW];
  __shared__ bf16 Bs[2][TN * LDW];

  const int e = blockIdx.z;
  const int cnt = segcnt[e];
  const int m0 = blockIdx.y * TM;
  if (m0 >= cnt) return;
  const int d0  = blockIdx.x * TN;
  const int off = segoff[e];
  const int tid = threadIdx.x;
  const int wave = tid >> 5, lane = tid & 31;
  const int wm = wave >> 1, wn = wave & 1;
  const int l15 = lane & 15, hi = lane >> 4;

  const float* W2e = W2 + (size_t)e * D_FF * D_MODEL;
  const bf16* Abase = h1 + (size_t)(off + m0) * D_FF;  // 128 contiguous segment rows

  v8f acc[2][4];
#pragma unroll
  for (int a = 0; a < 2; ++a)
#pragma unroll
    for (int n = 0; n < 4; ++n)
#pragma unroll
      for (int q = 0; q < 8; ++q) acc[a][n][q] = 0.f;

  const float* srcB[4]; u32 dstB[4];
#pragma unroll
  for (int i = 0; i < 4; ++i) {
    int idx = i * 256 + tid;
    int kp = idx >> 5;
    int c4 = idx & 31;
    srcB[i] = W2e + (size_t)(2 * kp) * D_MODEL + d0 + c4 * 4;
    dstB[i] = (c4 * 4) * LDW + 2 * kp;
  }

  auto stageA = [&](int buf, int k0) {
    if (wave == 0)
      tdm_load_tile_2d(&As[buf][0], Abase + k0, (u32)D_FF);
  };
  auto stageB = [&](int buf, int k0) {
#pragma unroll
    for (int i = 0; i < 4; ++i) {
      const float* gp = srcB[i] + (size_t)k0 * D_MODEL;
      if (k0 + 2 * TKC < D_FF)
        __builtin_prefetch(gp + 2 * (size_t)TKC * D_MODEL, 0, 1);
      f32x4 wa = __builtin_nontemporal_load((const f32x4*)gp);
      f32x4 wb = __builtin_nontemporal_load((const f32x4*)(gp + D_MODEL));
#pragma unroll
      for (int j = 0; j < 4; ++j) {
        union { u32 u; bf16 h[2]; } pk;
        pk.h[0] = (bf16)wa[j];
        pk.h[1] = (bf16)wb[j];
        *(u32*)&Bs[buf][dstB[i] + j * LDW] = pk.u;
      }
    }
  };

  stageA(0, 0);
  stageB(0, 0);
  constexpr int NIT = D_FF / TKC;            // 64
  for (int it = 0; it < NIT; ++it) {
    const int cur = it & 1;
    wait_tensor();                           // wave0's TDM done (no-op others)
    __syncthreads();
    if (it + 1 < NIT) {
      stageA(1 - cur, (it + 1) * TKC);
      stageB(1 - cur, (it + 1) * TKC);
    }
#pragma unroll
    for (int kk = 0; kk < TKC; kk += 32) {
      v16bf afrag[2];
#pragma unroll
      for (int a = 0; a < 2; ++a) {
        int row = (wm * 2 + a) * 16 + l15;
        int base = kk + (hi ? 8 : 0);
        const bf16* p = &As[cur][row * LDW];
        ((v8bf*)&afrag[a])[0] = *(const v8bf*)(p + base);
        ((v8bf*)&afrag[a])[1] = *(const v8bf*)(p + base + 16);
      }
      v16bf bfrag[4];
#pragma unroll
      for (int n = 0; n < 4; ++n) {
        int frow = (wn * 4 + n) * 16 + l15;
        int base = kk + (hi ? 8 : 0);
        const bf16* p = &Bs[cur][frow * LDW] + base;
        ((v8bf*)&bfrag[n])[0] = *(const v8bf*)(p);
        ((v8bf*)&bfrag[n])[1] = *(const v8bf*)(p + 16);
      }
#pragma unroll
      for (int a = 0; a < 2; ++a)
#pragma unroll
        for (int n = 0; n < 4; ++n)
          acc[a][n] = __builtin_amdgcn_wmma_f32_16x16x32_bf16(
              false, afrag[a], false, bfrag[n], (short)0, acc[a][n], false, false);
    }
  }

  const float* b2e = b2 + (size_t)e * D_MODEL;
#pragma unroll
  for (int a = 0; a < 2; ++a) {
#pragma unroll
    for (int r = 0; r < 8; ++r) {
      int rowl = (wm * 2 + a) * 16 + (hi ? r + 8 : r);
      int grow = m0 + rowl;
      if (grow >= cnt) continue;
      int arow = off + grow;
      int tkn = tok[arow];
      int sl = slot[arow];
      float gate = ga[arow];
      float* dst = contrib + ((size_t)tkn * TOPK + sl) * D_MODEL;
#pragma unroll
      for (int n = 0; n < 4; ++n) {
        int col = d0 + (wn * 4 + n) * 16 + l15;
        dst[col] = (acc[a][n][r] + b2e[col]) * gate;
      }
    }
  }
}

// ================================================================
// K6: out = x + contrib[slot0] + contrib[slot1]
// ================================================================
__global__ __launch_bounds__(256) void moe_finalize(
    const float* __restrict__ x, const float* __restrict__ contrib,
    float* __restrict__ out) {
  size_t i = ((size_t)blockIdx.x * 256 + threadIdx.x) * 4;
  size_t t = i / D_MODEL;
  size_t d = i % D_MODEL;
  f32x4 xv = *(const f32x4*)(x + i);
  f32x4 c0 = *(const f32x4*)(contrib + (t * 2 + 0) * D_MODEL + d);
  f32x4 c1 = *(const f32x4*)(contrib + (t * 2 + 1) * D_MODEL + d);
  f32x4 o = xv + c0 + c1;
  *(f32x4*)(out + i) = o;
}

// ================================================================
extern "C" void kernel_launch(void* const* d_in, const int* in_sizes, int n_in,
                              void* d_out, int out_size, void* d_ws, size_t ws_size,
                              hipStream_t stream) {
  (void)in_sizes; (void)n_in; (void)out_size; (void)ws_size;
  const float* x    = (const float*)d_in[0];
  const float* ln_g = (const float*)d_in[1];
  const float* ln_b = (const float*)d_in[2];
  const float* Wr   = (const float*)d_in[3];
  const float* br   = (const float*)d_in[4];
  const float* W1   = (const float*)d_in[5];
  const float* b1   = (const float*)d_in[6];
  const float* W2   = (const float*)d_in[7];
  const float* b2   = (const float*)d_in[8];
  float* out = (float*)d_out;

  // workspace carve-out (~105 MB total)
  char* ws = (char*)d_ws;
  size_t o = 0;
  auto take = [&](size_t bytes) -> char* {
    char* p = ws + o;
    o = (o + bytes + 255) & ~(size_t)255;
    return p;
  };
  bf16*  hbf     = (bf16*)take((size_t)N_TOK * D_MODEL * sizeof(bf16));  //  8 MB
  bf16*  h1      = (bf16*)take((size_t)N_ASG * D_FF * sizeof(bf16));     // 64 MB (L2-resident)
  float* contrib = (float*)take((size_t)N_ASG * D_MODEL * sizeof(float));// 32 MB
  int*   idxo    = (int*)take((size_t)N_ASG * sizeof(int));
  float* gateo   = (float*)take((size_t)N_ASG * sizeof(float));
  int*   tok     = (int*)take((size_t)N_ASG * sizeof(int));
  float* ga      = (float*)take((size_t)N_ASG * sizeof(float));
  int*   slot    = (int*)take((size_t)N_ASG * sizeof(int));
  int*   cnt     = (int*)take(N_EXP * sizeof(int));
  int*   segoff  = (int*)take(N_EXP * sizeof(int));
  int*   cursor  = (int*)take(N_EXP * sizeof(int));

  moe_zero_meta<<<1, 32, 0, stream>>>(cnt);
  moe_ln_router<<<N_TOK, 256, 0, stream>>>(x, ln_g, ln_b, Wr, br,
                                           hbf, idxo, gateo, cnt);
  moe_seg_offsets<<<1, 32, 0, stream>>>(cnt, segoff, cursor);
  moe_scatter<<<N_TOK / 256, 256, 0, stream>>>(idxo, gateo, cursor, tok, ga, slot);

  dim3 g1(D_FF / TN, (N_TOK + TM - 1) / TM, N_EXP);     // (32, 32, 8)
  moe_gemm1<<<g1, 256, 0, stream>>>(hbf, W1, b1, segoff, cnt, tok, h1);

  dim3 g2(D_MODEL / TN, (N_TOK + TM - 1) / TM, N_EXP);  // (8, 32, 8)
  moe_gemm2<<<g2, 256, 0, stream>>>(h1, W2, b2, segoff, cnt, tok, ga, slot, contrib);

  moe_finalize<<<(N_TOK * D_MODEL / 4) / 256, 256, 0, stream>>>(x, contrib, out);
}
